// SupervisedLoss_49709951484395
// MI455X (gfx1250) — matrix operand order repaired and checked by reading, not compiled, not run on hardware
//
#include <hip/hip_runtime.h>

// Soft-DTW supervised loss for MI455X (gfx1250, wave32, WMMA).
//
// Pipeline:
//   1) norm_kernel      : row norms ||x_i||^2 for outputs & targets (f32)
//   2) dist_wmma_kernel : D[b] = xn[i] + yn[j] - 2 * X·Y^T via v_wmma_f32_16x16x32_f16
//                         (f16 inputs, f32 accumulate), one 16x16 tile per wave
//   3) sdtw_kernel      : anti-diagonal wavefront DP, one workgroup per batch entry,
//                         rolling 3-row LDS buffer, one s_barrier per diagonal
//   4) reduce_kernel    : mean(r_xy - 0.5*(r_xx + r_yy)) + 1e-5
//
// ws layout (floats):
//   [0, 384*65536)        D matrices (~96 MB)
//   [+0, +65536)          row norms (outputs rows 0..32767, targets rows 32768..65535)
//   [+0, +384)            per-batch soft-DTW results

typedef __attribute__((ext_vector_type(16))) _Float16 v16h;
typedef __attribute__((ext_vector_type(8)))  float    v8f;

#define BATCH 128
#define SEQ   256
#define DIM   64
#define NB3   (3 * BATCH)      // 384 DP problems
#define DMAT  (SEQ * SEQ)      // 65536 elements per cost matrix
#define INFV  1e8f             // matches reference's finite "infinity"

// ---------------------------------------------------------------------------
// Kernel 1: per-row squared norms for both input tensors.
__global__ __launch_bounds__(256) void norm_kernel(
    const float* __restrict__ outs, const float* __restrict__ tgts,
    float* __restrict__ norms)
{
    int idx = blockIdx.x * blockDim.x + threadIdx.x;   // 0 .. 2*128*256-1
    const float* row = (idx < BATCH * SEQ)
                     ? (outs + (size_t)idx * DIM)
                     : (tgts + (size_t)(idx - BATCH * SEQ) * DIM);
    float s = 0.0f;
#pragma unroll
    for (int k = 0; k < DIM; k += 4) {
        float4 q = *(const float4*)(row + k);
        s += q.x * q.x + q.y * q.y + q.z * q.z + q.w * q.w;
    }
    norms[idx] = s;
}

// ---------------------------------------------------------------------------
// Kernel 2: batched cost matrices via WMMA. One wave = one 16x16 output tile.
// 384 batches * 256 tiles = 98304 waves; 8 waves per 256-thread block.
__global__ __launch_bounds__(256) void dist_wmma_kernel(
    const float* __restrict__ outs, const float* __restrict__ tgts,
    const float* __restrict__ norms, float* __restrict__ Dmat)
{
    const int lane = threadIdx.x & 31;
    const int wid  = blockIdx.x * (blockDim.x >> 5) + (threadIdx.x >> 5);
    const int bb   = wid >> 8;          // batch entry 0..383
    const int tile = wid & 255;
    const int tm   = tile >> 4;         // tile row
    const int tn   = tile & 15;         // tile col

    // Batch mapping: [0,128) -> (out,tgt); [128,256) -> (out,out); [256,384) -> (tgt,tgt)
    const int ob = (bb < BATCH) ? bb : (bb < 2 * BATCH ? bb - BATCH : bb - 2 * BATCH);
    const float* normO = norms;
    const float* normT = norms + BATCH * SEQ;
    const float *X, *Y, *xn, *yn;
    if (bb < BATCH) {
        X = outs + (size_t)ob * SEQ * DIM;  Y = tgts + (size_t)ob * SEQ * DIM;
        xn = normO + ob * SEQ;              yn = normT + ob * SEQ;
    } else if (bb < 2 * BATCH) {
        X = outs + (size_t)ob * SEQ * DIM;  Y = X;
        xn = normO + ob * SEQ;              yn = xn;
    } else {
        X = tgts + (size_t)ob * SEQ * DIM;  Y = X;
        xn = normT + ob * SEQ;              yn = xn;
    }

    const int half = lane >> 4;     // 0: lanes 0-15, 1: lanes 16-31
    const int l16  = lane & 15;
    const int am   = tm * 16 + l16; // A row owned by this lane (ISA 16-bit A layout)
    const int bn   = tn * 16 + l16; // B column (= Y row) owned by this lane

    v8f acc = {};
#pragma unroll
    for (int ks = 0; ks < DIM; ks += 32) {
        // A: 16x32 f16. VGPR v (pair of halves): K = ks + (v<4?0:16) + 8*half + 2*(v%4)
        v16h av;
#pragma unroll
        for (int v = 0; v < 8; ++v) {
            int kk = ks + ((v < 4) ? 0 : 16) + half * 8 + 2 * (v & 3);
            float2 p = *(const float2*)(X + (size_t)am * DIM + kk);
            av[2 * v]     = (_Float16)p.x;
            av[2 * v + 1] = (_Float16)p.y;
        }
        // B: 32x16 f16. Same K order as A: element e -> K = ks + (e<8?0:16) + 8*half + (e%8).
        v16h bv;
        int kb = ks + half * 8;
#pragma unroll
        for (int e = 0; e < 16; e += 4) {
            float4 q = *(const float4*)(Y + (size_t)bn * DIM + kb + e + ((e >= 8) ? 8 : 0));
            bv[e]     = (_Float16)q.x;
            bv[e + 1] = (_Float16)q.y;
            bv[e + 2] = (_Float16)q.z;
            bv[e + 3] = (_Float16)q.w;
        }
        acc = __builtin_amdgcn_wmma_f32_16x16x32_f16(
            /*neg_a=*/false, av, /*neg_b=*/false, bv,
            /*c_mod=*/(short)0, acc, /*reuse_a=*/false, /*reuse_b=*/false);
    }

    // C/D layout: VGPR r -> M = tm*16 + r + 8*half ; N = bn. D = xn + yn - 2*acc.
    float* out = Dmat + (size_t)bb * DMAT;
#pragma unroll
    for (int r = 0; r < 8; ++r) {
        int mo = tm * 16 + r + half * 8;
        out[(size_t)mo * SEQ + bn] = xn[mo] + yn[bn] - 2.0f * acc[r];
    }
}

// ---------------------------------------------------------------------------
// Kernel 3: soft-DTW wavefront DP. One 256-thread workgroup per batch entry.
// Thread t owns row i = t; diagonals advance with one barrier each.
__global__ __launch_bounds__(256) void sdtw_kernel(
    const float* __restrict__ Dmat, float* __restrict__ res)
{
    __shared__ float R[3][SEQ];
    const int t  = threadIdx.x;
    const int bb = blockIdx.x;
    const float* D = Dmat + (size_t)bb * DMAT;

    R[0][t] = INFV; R[1][t] = INFV; R[2][t] = INFV;
    __syncthreads();

    for (int d = 0; d < 2 * SEQ - 1; ++d) {
        const int j = d - t;
        const float* r1 = R[(d + 2) % 3];   // diagonal d-1
        const float* r2 = R[(d + 1) % 3];   // diagonal d-2
        float rn = INFV;
        if (j >= 0 && j < SEQ) {
            float a = (t == 0) ? ((d == 0) ? 0.0f : INFV) : r2[t - 1]; // R[i-1][j-1]
            float b = (t == 0) ? INFV : r1[t - 1];                      // R[i-1][j]
            float c = r1[t];                                            // R[i][j-1]
            float m = fminf(a, fminf(b, c));
            float s = __expf(m - a) + __expf(m - b) + __expf(m - c);    // gamma = 1
            rn = D[(size_t)t * SEQ + j] + (m - __logf(s));
        }
        R[d % 3][t] = rn;      // row d%3 holds data from d-3: dead, no intra-step race
        __syncthreads();       // single barrier separates step d reads/writes from d+1
    }

    if (t == 0) res[bb] = R[(2 * SEQ - 2) % 3][SEQ - 1];   // R[N-1][M-1]
}

// ---------------------------------------------------------------------------
// Kernel 4: final normalized mean.
__global__ __launch_bounds__(128) void reduce_kernel(
    const float* __restrict__ res, float* __restrict__ out)
{
    __shared__ float s[BATCH];
    int t = threadIdx.x;
    s[t] = res[t] - 0.5f * (res[BATCH + t] + res[2 * BATCH + t]);
    __syncthreads();
    for (int off = BATCH / 2; off > 0; off >>= 1) {
        if (t < off) s[t] += s[t + off];
        __syncthreads();
    }
    if (t == 0) out[0] = s[0] / (float)BATCH + 1e-5f;
}

// ---------------------------------------------------------------------------
extern "C" void kernel_launch(void* const* d_in, const int* in_sizes, int n_in,
                              void* d_out, int out_size, void* d_ws, size_t ws_size,
                              hipStream_t stream)
{
    const float* outs = (const float*)d_in[0];   // (128, 256, 64) f32
    const float* tgts = (const float*)d_in[1];   // (128, 256, 64) f32

    float* ws    = (float*)d_ws;
    float* Dmat  = ws;                               // 384 * 65536 floats (~96 MB)
    float* norms = ws + (size_t)NB3 * DMAT;          // 65536 floats
    float* res   = norms + 2 * BATCH * SEQ;          // 384 floats

    // 1) row norms: 65536 rows, one thread each
    norm_kernel<<<(2 * BATCH * SEQ) / 256, 256, 0, stream>>>(outs, tgts, norms);

    // 2) cost matrices: 98304 waves, 8 waves per block -> 12288 blocks
    dist_wmma_kernel<<<(NB3 * 256) / 8, 256, 0, stream>>>(outs, tgts, norms, Dmat);

    // 3) DP wavefront: one workgroup per batch entry
    sdtw_kernel<<<NB3, SEQ, 0, stream>>>(Dmat, res);

    // 4) scalar reduction
    reduce_kernel<<<1, BATCH, 0, stream>>>(res, (float*)d_out);
}
